// TransitionDown_18923625906662
// MI455X (gfx1250) — matrix operand order repaired and checked by reading, not compiled, not run on hardware
//
#include <hip/hip_runtime.h>

#define NPTS 8192
#define MPTS 2048
#define BATCH 4
#define CIN 32
#define COUT 64
#define KNN 16

typedef __attribute__((ext_vector_type(16))) _Float16 v16h;
typedef __attribute__((ext_vector_type(8)))  _Float16 v8h;
typedef __attribute__((ext_vector_type(8)))  float    v8f;

// ---------------------------------------------------------------------------
// Kernel 1: furthest point sampling. One block per batch, 1024 threads,
// each thread owns 8 points cached in registers. 2047 serial argmax steps.
// Writes p2 (B,M,3) directly into d_out.
// ---------------------------------------------------------------------------
__global__ __launch_bounds__(1024) void fps_kernel(const float* __restrict__ p1,
                                                   float* __restrict__ p2) {
  const int b = blockIdx.x;
  const int tid = threadIdx.x;
  const int lane = tid & 31, wave = tid >> 5;
  __shared__ float rv[32];
  __shared__ int   ri[32];
  __shared__ float lastc[3];

  float px[8], py[8], pz[8], dist[8];
#pragma unroll
  for (int j = 0; j < 8; ++j) {
    int n = tid + j * 1024;
    const float* pp = p1 + ((size_t)b * NPTS + n) * 3;
    px[j] = pp[0]; py[j] = pp[1]; pz[j] = pp[2];
    dist[j] = 1e10f;
  }
  if (tid == 0) {
    const float* pp = p1 + (size_t)b * NPTS * 3;
    lastc[0] = pp[0]; lastc[1] = pp[1]; lastc[2] = pp[2];
    float* o = p2 + (size_t)b * MPTS * 3;
    o[0] = pp[0]; o[1] = pp[1]; o[2] = pp[2];
  }
  __syncthreads();

  for (int it = 1; it < MPTS; ++it) {
    const float lx = lastc[0], ly = lastc[1], lz = lastc[2];
    float bv = -1.f; int bi = 0x7fffffff;
#pragma unroll
    for (int j = 0; j < 8; ++j) {
      float dx = px[j] - lx, dy = py[j] - ly, dz = pz[j] - lz;
      float d = dx * dx + dy * dy + dz * dz;
      d = fminf(dist[j], d);
      dist[j] = d;
      int n = tid + j * 1024;
      if (d > bv || (d == bv && n < bi)) { bv = d; bi = n; }
    }
#pragma unroll
    for (int off = 16; off; off >>= 1) {
      float ov = __shfl_down(bv, off);
      int   oi = __shfl_down(bi, off);
      if (ov > bv || (ov == bv && oi < bi)) { bv = ov; bi = oi; }
    }
    if (lane == 0) { rv[wave] = bv; ri[wave] = bi; }
    __syncthreads();
    if (wave == 0) {
      bv = rv[lane]; bi = ri[lane];
#pragma unroll
      for (int off = 16; off; off >>= 1) {
        float ov = __shfl_down(bv, off);
        int   oi = __shfl_down(bi, off);
        if (ov > bv || (ov == bv && oi < bi)) { bv = ov; bi = oi; }
      }
      if (lane == 0) {
        const float* pp = p1 + ((size_t)b * NPTS + bi) * 3;
        float sx = pp[0], sy = pp[1], sz = pp[2];
        lastc[0] = sx; lastc[1] = sy; lastc[2] = sz;
        float* o = p2 + ((size_t)b * MPTS + it) * 3;
        o[0] = sx; o[1] = sy; o[2] = sz;
      }
    }
    __syncthreads();
  }
}

// ---------------------------------------------------------------------------
// Kernel 2: kNN. One wave per query. Per-lane register top-16 over 256
// candidates, then 16-round wave-wide merge via butterfly shuffles.
// ---------------------------------------------------------------------------
__global__ __launch_bounds__(256) void knn_kernel(const float* __restrict__ p1,
                                                  const float* __restrict__ p2,
                                                  int* __restrict__ nn) {
  const int lane = threadIdx.x & 31;
  const int q = blockIdx.x * 8 + (threadIdx.x >> 5);
  const int b = q >> 11;
  const float qx = p2[q * 3 + 0], qy = p2[q * 3 + 1], qz = p2[q * 3 + 2];

  float d16[16]; int i16[16];
#pragma unroll
  for (int s = 0; s < 16; ++s) { d16[s] = 3.4e38f; i16[s] = 0x7fffffff; }
  float worst = 3.4e38f; int wslot = 0;

  const float* pb = p1 + (size_t)b * NPTS * 3;
  for (int i = 0; i < NPTS / 32; ++i) {
    int n = lane + i * 32;
    float dx = pb[n * 3 + 0] - qx, dy = pb[n * 3 + 1] - qy, dz = pb[n * 3 + 2] - qz;
    float d = dx * dx + dy * dy + dz * dz;
    if (d < worst) {
#pragma unroll
      for (int s = 0; s < 16; ++s) if (s == wslot) { d16[s] = d; i16[s] = n; }
      worst = -1.f;
#pragma unroll
      for (int s = 0; s < 16; ++s) if (d16[s] > worst) { worst = d16[s]; wslot = s; }
    }
  }
  // local sort ascending (unrolled -> stays in registers)
#pragma unroll
  for (int i = 0; i < 15; ++i)
#pragma unroll
    for (int j = 0; j < 15; ++j)
      if (d16[j] > d16[j + 1] || (d16[j] == d16[j + 1] && i16[j] > i16[j + 1])) {
        float td = d16[j]; d16[j] = d16[j + 1]; d16[j + 1] = td;
        int   ti = i16[j]; i16[j] = i16[j + 1]; i16[j + 1] = ti;
      }
  // wave merge: 16 rounds of min over 32 sorted-list heads
  int hp = 0, mysel = 0;
  for (int r = 0; r < 16; ++r) {
    float hv = 3.4e38f; int hi = 0x7fffffff;
#pragma unroll
    for (int s = 0; s < 16; ++s) if (s == hp) { hv = d16[s]; hi = i16[s]; }
    int hl = lane;
#pragma unroll
    for (int off = 16; off; off >>= 1) {
      float ov = __shfl_xor(hv, off);
      int   oi = __shfl_xor(hi, off);
      int   ol = __shfl_xor(hl, off);
      if (ov < hv || (ov == hv && oi < hi)) { hv = ov; hi = oi; hl = ol; }
    }
    if (lane == hl) hp++;
    if (lane == r)  mysel = hi;
  }
  if (lane < 16) nn[q * 16 + lane] = mysel;
}

// ---------------------------------------------------------------------------
// Kernel 3: gather + MLP(35->64->64) + max-pool over K, WMMA per wave.
// One wave = one query: its K=16 neighbors are the 16 rows of the GEMM tile.
// ---------------------------------------------------------------------------
__global__ __launch_bounds__(256) void mlp_kernel(
    const float* __restrict__ p1, const float* __restrict__ x,
    const float* __restrict__ w1, const float* __restrict__ g1,
    const float* __restrict__ b1, const float* __restrict__ m1,
    const float* __restrict__ v1,
    const float* __restrict__ w2, const float* __restrict__ g2,
    const float* __restrict__ b2, const float* __restrict__ m2,
    const float* __restrict__ v2,
    const int* __restrict__ nn, const float* __restrict__ p2,
    float* __restrict__ y) {
  // 16 B-operand sets (8 per layer: 4 N-tiles x 2 K-chunks), per-lane dwords
  __shared__ unsigned int wB[16][32][8];   // 16 KB
  __shared__ _Float16 hbuf[8][16][64];     // per-wave 16x64 f16 tile, 16 KB
  const int tid = threadIdx.x;

  // Build weight operand image in a B-matrix (32x16 f16) lane layout with contiguous k (same order as ldA):
  // lane -> column n = 16*t + (lane&15); VGPR v holds K = 32c + 16*(lane>>4) + 2v + {0,1}
  for (int r = tid; r < 16 * 32 * 8; r += 256) {
    int set = r >> 8, rem = r & 255, ln = rem >> 3, v = rem & 7;
    const float* wsrc; int kdim, n, kb;
    if (set < 8) { int t = set >> 1, c = set & 1;
      n = t * 16 + (ln & 15); kb = 32 * c + 16 * (ln >> 4) + 2 * v;
      wsrc = w1; kdim = 35;
    } else { int s2 = set - 8, t = s2 >> 1, c = s2 & 1;
      n = t * 16 + (ln & 15); kb = 32 * c + 16 * (ln >> 4) + 2 * v;
      wsrc = w2; kdim = 64;
    }
    float f0 = (kb     < kdim) ? wsrc[n * kdim + kb]     : 0.f;
    float f1 = (kb + 1 < kdim) ? wsrc[n * kdim + kb + 1] : 0.f;
    union { _Float16 h[2]; unsigned int u; } pk;
    pk.h[0] = (_Float16)f0; pk.h[1] = (_Float16)f1;
    wB[set][ln][v] = pk.u;
  }
  __syncthreads();

  const int w = tid >> 5, lane = tid & 31;
  const int ln15 = lane & 15, half = lane >> 4;
  const int q = blockIdx.x * 8 + w;
  const int b = q >> 11, m = q & 2047;

  // Stage gathered features (zero-padded 35 -> 64) as f16 rows in LDS.
  const int nnr = nn[q * 16 + ln15];
  const float qx = p2[q * 3 + 0], qy = p2[q * 3 + 1], qz = p2[q * 3 + 2];
  if (half == 0) {
    const float* pp = p1 + ((size_t)b * NPTS + nnr) * 3;
    hbuf[w][ln15][0] = (_Float16)(pp[0] - qx);
    hbuf[w][ln15][1] = (_Float16)(pp[1] - qy);
    hbuf[w][ln15][2] = (_Float16)(pp[2] - qz);
#pragma unroll
    for (int c = 0; c < 16; ++c)
      hbuf[w][ln15][3 + c] = (_Float16)x[((size_t)b * CIN + c) * NPTS + nnr];
  } else {
#pragma unroll
    for (int c = 16; c < 32; ++c)
      hbuf[w][ln15][3 + c] = (_Float16)x[((size_t)b * CIN + c) * NPTS + nnr];
#pragma unroll
    for (int f = 35; f < 64; ++f) hbuf[w][ln15][f] = (_Float16)0.f;
  }
  __syncthreads();

  const int b0 = half * 16;  // A layout: lane holds row ln15, K base 0 or 16
  auto ldA = [&](int kbase) -> v16h {
    v8h lo = *reinterpret_cast<const v8h*>(&hbuf[w][ln15][kbase + b0]);
    v8h hi = *reinterpret_cast<const v8h*>(&hbuf[w][ln15][kbase + 8 + b0]);
    v16h a;
#pragma unroll
    for (int i = 0; i < 8; ++i) { a[i] = lo[i]; a[i + 8] = hi[i]; }
    return a;
  };
  auto ldB = [&](int set) -> v16h {
    union { uint4 u[2]; v16h h; } t;
    t.u[0] = *reinterpret_cast<const uint4*>(&wB[set][lane][0]);
    t.u[1] = *reinterpret_cast<const uint4*>(&wB[set][lane][4]);
    return t.h;
  };

  // ----- layer 1: (16x64) = (16x64pad) x (64x64pad of w1^T), 8 WMMAs -----
  v16h a0 = ldA(0), a1 = ldA(32);
  v8f acc[4];
#pragma unroll
  for (int t = 0; t < 4; ++t) {
    v8f c = {};
    c = __builtin_amdgcn_wmma_f32_16x16x32_f16(false, a0, false, ldB(t * 2 + 0),
                                               (short)0, c, false, false);
    c = __builtin_amdgcn_wmma_f32_16x16x32_f16(false, a1, false, ldB(t * 2 + 1),
                                               (short)0, c, false, false);
    acc[t] = c;
  }
  // BN + ReLU; each lane's 8 accumulators share channel ch = 16t + ln15.
#pragma unroll
  for (int t = 0; t < 4; ++t) {
    int ch = t * 16 + ln15;
    float s  = g1[ch] * rsqrtf(v1[ch] + 1e-5f);
    float bb = b1[ch] - m1[ch] * s;
#pragma unroll
    for (int v = 0; v < 8; ++v) {
      float f = fmaxf(acc[t][v] * s + bb, 0.f);
      hbuf[w][half ? v + 8 : v][ch] = (_Float16)f;  // C row = v (+8 hi half)
    }
  }
  __syncthreads();

  // ----- layer 2: (16x64) = (16x64) x (64x64 of w2^T), 8 WMMAs -----
  a0 = ldA(0); a1 = ldA(32);
#pragma unroll
  for (int t = 0; t < 4; ++t) {
    v8f c = {};
    c = __builtin_amdgcn_wmma_f32_16x16x32_f16(false, a0, false, ldB(8 + t * 2 + 0),
                                               (short)0, c, false, false);
    c = __builtin_amdgcn_wmma_f32_16x16x32_f16(false, a1, false, ldB(8 + t * 2 + 1),
                                               (short)0, c, false, false);
    acc[t] = c;
  }
  // BN + ReLU + max over the 16 rows (K neighbors), write y[b][ch][m].
#pragma unroll
  for (int t = 0; t < 4; ++t) {
    int ch = t * 16 + ln15;
    float s  = g2[ch] * rsqrtf(v2[ch] + 1e-5f);
    float bb = b2[ch] - m2[ch] * s;
    float mx = -3.4e38f;
#pragma unroll
    for (int v = 0; v < 8; ++v) mx = fmaxf(mx, fmaxf(acc[t][v] * s + bb, 0.f));
    mx = fmaxf(mx, __shfl_xor(mx, 16));  // combine row halves
    if (lane < 16)
      y[((size_t)b * COUT + ch) * MPTS + m] = mx;
  }
}

// ---------------------------------------------------------------------------
extern "C" void kernel_launch(void* const* d_in, const int* in_sizes, int n_in,
                              void* d_out, int out_size, void* d_ws, size_t ws_size,
                              hipStream_t stream) {
  const float* p1 = (const float*)d_in[0];
  const float* x  = (const float*)d_in[1];
  const float* w1 = (const float*)d_in[2];
  const float* g1 = (const float*)d_in[3];
  const float* b1 = (const float*)d_in[4];
  const float* m1 = (const float*)d_in[5];
  const float* v1 = (const float*)d_in[6];
  const float* w2 = (const float*)d_in[7];
  const float* g2 = (const float*)d_in[8];
  const float* b2 = (const float*)d_in[9];
  const float* m2 = (const float*)d_in[10];
  const float* v2 = (const float*)d_in[11];

  float* out = (float*)d_out;
  float* p2  = out;                          // (B,M,3) = 24576 floats
  float* y   = out + (size_t)BATCH * MPTS * 3;  // (B,COUT,M)
  int*   nn  = (int*)d_ws;                   // (B,M,K) ints = 512 KB scratch

  fps_kernel<<<BATCH, 1024, 0, stream>>>(p1, p2);
  knn_kernel<<<(BATCH * MPTS) / 8, 256, 0, stream>>>(p1, p2, nn);
  mlp_kernel<<<(BATCH * MPTS) / 8, 256, 0, stream>>>(p1, x, w1, g1, b1, m1, v1,
                                                     w2, g2, b2, m2, v2, nn, p2, y);
}
